// ExRestSelfAtten_87711822119181
// MI455X (gfx1250) — matrix operand order repaired and checked by reading, not compiled, not run on hardware
//
#include <hip/hip_runtime.h>
#include <hip/hip_bf16.h>

// ---------------------------------------------------------------------------
// ExRestSelfAtten on MI455X (gfx1250, wave32, WMMA f32_16x16x32_f16,
// async global->LDS staging).
//
//  h   = relu(x @ W1 + b1)                      (16384 x 256 x 256 GEMM)
//  Q   = h @ Wq ; Kfull = h @ Wk ; Vfull = h @ Wv   (fused in kernel 1)
//  logits[s,w] = Q[s] . Kfull[s+16-w] / 16   (zero rows outside [0,S) -> logit 0)
//  attn = softmax over w=0..32 ; attended = sum attn * Vfull[s+16-w]
//  hid = relu(attended @ Wh + bh) ; out = hid @ Wo + bo  (fused in kernel 2)
//
// Workspace layout (halves):
//   [0)            5 packed f16 weights, 65536 halves each (WMMA B-frag order)
//   [5*65536)      Q  (16384*256 f16)
//   [+4194304)     K  (16384*256 f16)
//   [+4194304)     V  (16384*256 f16)
// Total = 25,821,184 bytes.
// ---------------------------------------------------------------------------

typedef __attribute__((ext_vector_type(16))) _Float16 v16h;
typedef __attribute__((ext_vector_type(8)))  _Float16 v8h;
typedef __attribute__((ext_vector_type(8)))  float    v8f;

#define Bsz   8
#define Ssz   2048
#define Dsz   256
#define Hsz   256
#define Asz   16
#define Osz   2
#define Wwin  33
#define LDST  264              // LDS row stride in halves (528 B, 16B aligned)
#define ROWS  16               // rows per block tile
#define HALO  48               // ROWS + 2*Asz

__device__ __forceinline__ v8f wmma16(v16h a, v16h b, v8f c) {
  // D = A(16x32 f16) * B(32x16 f16) + C(16x16 f32)
  return __builtin_amdgcn_wmma_f32_16x16x32_f16(false, a, false, b,
                                                (short)0, c, false, false);
}

// A-frag (16x32 f16) from LDS tile at rows 0..15, K-block kb (ISA 7.12.2).
__device__ __forceinline__ v16h load_a_frag(const _Float16* lds, int kb) {
  const int lane = threadIdx.x & 31;
  const int m    = lane & 15;
  const int koff = (lane >> 4) << 3;              // lanes 16-31: +8 in K
  const _Float16* p = lds + m * LDST + kb * 32 + koff;
  v8h lo = *(const v8h*)p;                        // K = base .. base+7
  v8h hi = *(const v8h*)(p + 16);                 // K = base+16 .. base+23
  v16h a;
#pragma unroll
  for (int i = 0; i < 8; ++i) { a[i] = lo[i]; a[i + 8] = hi[i]; }
  return a;
}

// B-frag (32x16 f16): pre-packed so each lane reads 16 contiguous halves.
__device__ __forceinline__ v16h load_b_frag(const _Float16* wp, int nb, int kb) {
  const int lane = threadIdx.x & 31;
  return *(const v16h*)(wp + (size_t)((((nb << 3) + kb) << 5) + lane) * 16);
}

// C/D frag (16x16 f32) scatter helpers (ISA 7.12.2 32-bit C/D layout).
__device__ __forceinline__ void store_d_f16_lds(_Float16* lds, int col0, v8f acc) {
  const int lane = threadIdx.x & 31;
  const int col   = col0 + (lane & 15);
  const int rbase = (lane >> 4) << 3;
#pragma unroll
  for (int i = 0; i < 8; ++i) lds[(rbase + i) * LDST + col] = (_Float16)acc[i];
}

__device__ __forceinline__ void store_d_f16_glb(_Float16* g, size_t row0, int col0, v8f acc) {
  const int lane = threadIdx.x & 31;
  const int col   = col0 + (lane & 15);
  const size_t rb = row0 + ((lane >> 4) << 3);
#pragma unroll
  for (int i = 0; i < 8; ++i) g[(rb + i) * Hsz + col] = (_Float16)acc[i];
}

__device__ __forceinline__ float wave_max(float v) {
#pragma unroll
  for (int o = 16; o; o >>= 1) v = fmaxf(v, __shfl_xor(v, o));
  return v;
}
__device__ __forceinline__ float wave_sum(float v) {
#pragma unroll
  for (int o = 16; o; o >>= 1) v += __shfl_xor(v, o);
  return v;
}

// ---------------------------------------------------------------------------
// Async global->LDS copy of one 512B row (wave-wide: 32 lanes x 16B).
// Must be executed with all 32 lanes of the wave active (row id wave-uniform).
// Tracked by ASYNCcnt; pair with async_wait_all() before the barrier.
// ---------------------------------------------------------------------------
__device__ __forceinline__ void async_copy_row512(const _Float16* g, _Float16* lds) {
  const int lane = threadIdx.x & 31;
  unsigned           laddr = (unsigned)(unsigned long long)lds + (unsigned)(lane * 16);
  unsigned long long gaddr = (unsigned long long)g + (unsigned long long)(lane * 16);
  asm volatile("global_load_async_to_lds_b128 %0, %1, off"
               :: "v"(laddr), "v"(gaddr) : "memory");
}

__device__ __forceinline__ void async_wait_all() {
  asm volatile("s_wait_asynccnt 0x0" ::: "memory");
}

// Zero one 512B LDS row (wave-wide).
__device__ __forceinline__ void zero_row512(_Float16* lds) {
  const int lane = threadIdx.x & 31;
  ((uint4*)lds)[lane] = uint4{0u, 0u, 0u, 0u};
}

// ---------------------------------------------------------------------------
// Kernel P: pack a 256x256 f32 weight into f16 WMMA B-fragment order.
// packed[((nb*8+kb)*32 + L)*16 + j] = W[kb*32 + (j<8 ? 8*(L>>4)+j : 16+8*(L>>4)+(j-8))][nb*16 + (L&15)]
// ---------------------------------------------------------------------------
__global__ void __launch_bounds__(256) pack_w_kernel(const float* __restrict__ src,
                                                     _Float16* __restrict__ dst) {
  const int p  = blockIdx.x * 256 + threadIdx.x;    // 65536 elements
  const int j  = p & 15;
  const int L  = (p >> 4) & 31;
  const int kb = (p >> 9) & 7;
  const int nb = p >> 12;
  const int kk = (j < 8) ? (((L >> 4) << 3) + j) : (16 + ((L >> 4) << 3) + (j - 8));
  const int nn = L & 15;
  dst[p] = (_Float16)src[(kb * 32 + kk) * Hsz + nb * 16 + nn];
}

// ---------------------------------------------------------------------------
// Kernel 1: per 16-row tile:  x -> h=relu(x@W1+b1) (LDS) -> Q/K/V (f16, ws)
// ---------------------------------------------------------------------------
__global__ void __launch_bounds__(256) qkv_kernel(
    const float* __restrict__ x, const float* __restrict__ b1,
    const _Float16* __restrict__ W1p, const _Float16* __restrict__ Wqp,
    const _Float16* __restrict__ Wkp, const _Float16* __restrict__ Wvp,
    _Float16* __restrict__ Qw, _Float16* __restrict__ Kw, _Float16* __restrict__ Vw) {
  __shared__ _Float16 xbuf[ROWS * LDST];
  __shared__ _Float16 hbuf[ROWS * LDST];

  const size_t g0 = (size_t)blockIdx.x * ROWS;      // flat (b*S+s) row base
  const int t = threadIdx.x;
  __builtin_prefetch(W1p, 0, 2);                    // global_prefetch_b8

  { // stage x tile as f16 (needs f32->f16 convert, so no async path here)
    const int r = t >> 4, c0 = (t & 15) * 16;
    const float* src = x + (g0 + r) * Dsz + c0;
#pragma unroll
    for (int i = 0; i < 16; ++i) xbuf[r * LDST + c0 + i] = (_Float16)src[i];
  }
  __syncthreads();

  const int wv = t >> 5, lane = t & 31;
  const int nb0 = wv * 2, nb1 = nb0 + 1;

  { // GEMM1: h = relu(x @ W1 + b1)
    v8f a0 = {}, a1 = {};
#pragma unroll
    for (int kb = 0; kb < 8; ++kb) {
      v16h af = load_a_frag(xbuf, kb);
      a0 = wmma16(af, load_b_frag(W1p, nb0, kb), a0);
      a1 = wmma16(af, load_b_frag(W1p, nb1, kb), a1);
    }
    const float bi0 = b1[nb0 * 16 + (lane & 15)];
    const float bi1 = b1[nb1 * 16 + (lane & 15)];
#pragma unroll
    for (int i = 0; i < 8; ++i) { a0[i] = fmaxf(a0[i] + bi0, 0.f);
                                  a1[i] = fmaxf(a1[i] + bi1, 0.f); }
    store_d_f16_lds(hbuf, nb0 * 16, a0);
    store_d_f16_lds(hbuf, nb1 * 16, a1);
  }
  __syncthreads();

  { // GEMM2: Q/K/V = h @ {Wq,Wk,Wv}, shared A-frags
    v8f q0 = {}, q1 = {}, k0 = {}, k1 = {}, v0 = {}, v1 = {};
#pragma unroll
    for (int kb = 0; kb < 8; ++kb) {
      v16h af = load_a_frag(hbuf, kb);
      q0 = wmma16(af, load_b_frag(Wqp, nb0, kb), q0);
      q1 = wmma16(af, load_b_frag(Wqp, nb1, kb), q1);
      k0 = wmma16(af, load_b_frag(Wkp, nb0, kb), k0);
      k1 = wmma16(af, load_b_frag(Wkp, nb1, kb), k1);
      v0 = wmma16(af, load_b_frag(Wvp, nb0, kb), v0);
      v1 = wmma16(af, load_b_frag(Wvp, nb1, kb), v1);
    }
    store_d_f16_glb(Qw, g0, nb0 * 16, q0);  store_d_f16_glb(Qw, g0, nb1 * 16, q1);
    store_d_f16_glb(Kw, g0, nb0 * 16, k0);  store_d_f16_glb(Kw, g0, nb1 * 16, k1);
    store_d_f16_glb(Vw, g0, nb0 * 16, v0);  store_d_f16_glb(Vw, g0, nb1 * 16, v1);
  }
}

// ---------------------------------------------------------------------------
// Kernel 2: per 16-row tile: banded attention + hid=relu(att@Wh+bh) + out.
// Q/K/V staged via GLOBAL_LOAD_ASYNC_TO_LDS_B128 (ASYNCcnt), OOB halo rows
// zero-filled with regular DS stores. Qbuf reused for attended (f16),
// Vbuf rows 0..15 reused for hid (f16).
// ---------------------------------------------------------------------------
__device__ __forceinline__ float dot256(const _Float16* qrow, const _Float16* krow) {
  const v8h* qp = (const v8h*)qrow;
  const v8h* kp = (const v8h*)krow;
  float s = 0.f;
#pragma unroll
  for (int i = 0; i < 32; ++i) {
    v8h q = qp[i], k = kp[i];
#pragma unroll
    for (int u = 0; u < 8; ++u) s += (float)q[u] * (float)k[u];
  }
  return s;
}

__global__ void __launch_bounds__(256) attn_kernel(
    const _Float16* __restrict__ Qw, const _Float16* __restrict__ Kw,
    const _Float16* __restrict__ Vw, const _Float16* __restrict__ Whp,
    const float* __restrict__ bh, const float* __restrict__ Wo,
    const float* __restrict__ bo, float* __restrict__ out) {
  __shared__ _Float16 Kbuf[HALO * LDST];
  __shared__ _Float16 Vbuf[HALO * LDST];
  __shared__ _Float16 Qbuf[ROWS * LDST];
  __shared__ float    attnw[ROWS][36];

  const int tilesPerBatch = Ssz / ROWS;             // 128
  const int b  = blockIdx.x / tilesPerBatch;
  const int s0 = (blockIdx.x % tilesPerBatch) * ROWS;
  const size_t g0 = (size_t)b * Ssz + s0;
  const int t = threadIdx.x;
  const int wv = t >> 5, lane = t & 31;

  // ---- async staging: 112 row-copies (16 Q + 48 K + 48 V), 512B each.
  // Row/task id is wave-uniform; lanes only differ in the 16B column chunk,
  // so each async instruction issues with all 32 lanes active.
  for (int task = wv; task < 16 + 2 * HALO; task += 8) {
    if (task < 16) {                               // Q row
      async_copy_row512(Qw + (g0 + task) * Hsz, Qbuf + task * LDST);
    } else if (task < 16 + HALO) {                 // K halo row
      const int j = task - 16;
      const int s = s0 + j - Asz;
      if (s >= 0 && s < Ssz)
        async_copy_row512(Kw + ((size_t)b * Ssz + s) * Hsz, Kbuf + j * LDST);
      else
        zero_row512(Kbuf + j * LDST);
    } else {                                       // V halo row
      const int j = task - 16 - HALO;
      const int s = s0 + j - Asz;
      if (s >= 0 && s < Ssz)
        async_copy_row512(Vw + ((size_t)b * Ssz + s) * Hsz, Vbuf + j * LDST);
      else
        zero_row512(Vbuf + j * LDST);
    }
  }
  async_wait_all();          // ASYNCcnt == 0 for this wave's copies
  __syncthreads();           // (compiler also waits DScnt for the zero-fills)

  // logits + softmax: wave handles rows wv and wv+8; lane l owns window w=l,
  // lane-uniform extra pass for w=32 (neighbor j = r). neighbor j = r+32-w.
#pragma unroll
  for (int rr = 0; rr < 2; ++rr) {
    const int r = wv + rr * 8;
    const float lg   = dot256(Qbuf + r * LDST, Kbuf + (r + 32 - lane) * LDST) * 0.0625f;
    const float lg32 = dot256(Qbuf + r * LDST, Kbuf + r * LDST) * 0.0625f;
    float m = fmaxf(wave_max(lg), lg32);
    const float e   = __expf(lg - m);
    const float e32 = __expf(lg32 - m);
    const float den = wave_sum(e) + e32;
    attnw[r][lane] = e / den;
    if (lane == 0) attnw[r][32] = e32 / den;
  }
  __syncthreads();   // Q/K done; Qbuf free for reuse

  // attended = sum_w attn[w] * V[r+32-w]; lane owns 8 channels; write f16->Qbuf
#pragma unroll
  for (int rr = 0; rr < 2; ++rr) {
    const int r = wv + rr * 8;
    float acc[8] = {0.f, 0.f, 0.f, 0.f, 0.f, 0.f, 0.f, 0.f};
#pragma unroll 4
    for (int w = 0; w < Wwin; ++w) {
      const float aw = attnw[r][w];
      const _Float16* vrow = Vbuf + (r + 32 - w) * LDST + lane * 8;
#pragma unroll
      for (int u = 0; u < 8; ++u) acc[u] += aw * (float)vrow[u];
    }
    _Float16* dst = Qbuf + r * LDST + lane * 8;
#pragma unroll
    for (int u = 0; u < 8; ++u) dst[u] = (_Float16)acc[u];
  }
  __syncthreads();   // attended staged in Qbuf; GEMM3 writes Vbuf rows 0..15

  { // GEMM3: hid = relu(attended @ Wh + bh) -> Vbuf rows 0..15 (f16)
    const int nb0 = wv * 2, nb1 = nb0 + 1;
    v8f a0 = {}, a1 = {};
#pragma unroll
    for (int kb = 0; kb < 8; ++kb) {
      v16h af = load_a_frag(Qbuf, kb);
      a0 = wmma16(af, load_b_frag(Whp, nb0, kb), a0);
      a1 = wmma16(af, load_b_frag(Whp, nb1, kb), a1);
    }
    const float bi0 = bh[nb0 * 16 + (lane & 15)];
    const float bi1 = bh[nb1 * 16 + (lane & 15)];
#pragma unroll
    for (int i = 0; i < 8; ++i) { a0[i] = fmaxf(a0[i] + bi0, 0.f);
                                  a1[i] = fmaxf(a1[i] + bi1, 0.f); }
    store_d_f16_lds(Vbuf, nb0 * 16, a0);
    store_d_f16_lds(Vbuf, nb1 * 16, a1);
  }
  __syncthreads();

  // out = hid @ Wo + bo  (16 rows x 2 cols) — wave 0 only
  if (t < 32) {
    const int r = t & 15, o = t >> 4;
    const _Float16* hrow = Vbuf + r * LDST;
    float s = 0.f;
#pragma unroll 8
    for (int c = 0; c < Hsz; ++c) s += (float)hrow[c] * Wo[c * Osz + o];
    out[(g0 + r) * Osz + o] = s + bo[o];
  }
}

// ---------------------------------------------------------------------------
extern "C" void kernel_launch(void* const* d_in, const int* in_sizes, int n_in,
                              void* d_out, int out_size, void* d_ws, size_t ws_size,
                              hipStream_t stream) {
  (void)in_sizes; (void)n_in; (void)out_size; (void)ws_size;
  const float* x  = (const float*)d_in[0];
  const float* W1 = (const float*)d_in[1];
  const float* b1 = (const float*)d_in[2];
  const float* Wq = (const float*)d_in[3];
  const float* Wk = (const float*)d_in[4];
  const float* Wv = (const float*)d_in[5];
  const float* Wh = (const float*)d_in[6];
  const float* bh = (const float*)d_in[7];
  const float* Wo = (const float*)d_in[8];
  const float* bo = (const float*)d_in[9];
  float* out = (float*)d_out;

  const size_t WPACK = (size_t)Dsz * Hsz;          // 65536 halves per matrix
  const size_t ACT   = (size_t)Bsz * Ssz * Hsz;    // 4194304 halves
  _Float16* wsH = (_Float16*)d_ws;                 // needs ~25.9 MB
  _Float16* W1p = wsH + 0 * WPACK;
  _Float16* Wqp = wsH + 1 * WPACK;
  _Float16* Wkp = wsH + 2 * WPACK;
  _Float16* Wvp = wsH + 3 * WPACK;
  _Float16* Whp = wsH + 4 * WPACK;
  _Float16* Qw  = wsH + 5 * WPACK;
  _Float16* Kw  = Qw + ACT;
  _Float16* Vw  = Kw + ACT;

  pack_w_kernel<<<256, 256, 0, stream>>>(W1, W1p);
  pack_w_kernel<<<256, 256, 0, stream>>>(Wq, Wqp);
  pack_w_kernel<<<256, 256, 0, stream>>>(Wk, Wkp);
  pack_w_kernel<<<256, 256, 0, stream>>>(Wv, Wvp);
  pack_w_kernel<<<256, 256, 0, stream>>>(Wh, Whp);

  const int nTiles = (Bsz * Ssz) / ROWS;           // 1024 blocks
  qkv_kernel<<<nTiles, 256, 0, stream>>>(x, b1, W1p, Wqp, Wkp, Wvp, Qw, Kw, Vw);
  attn_kernel<<<nTiles, 256, 0, stream>>>(Qw, Kw, Vw, Whp, bh, Wo, bo, out);
}
